// SpectralProxLayer_85177791414402
// MI455X (gfx1250) — matrix operand order repaired and checked by reading, not compiled, not run on hardware
//
#include <hip/hip_runtime.h>
#include <hip/hip_bf16.h>
#include <cstdint>

// ---------------------------------------------------------------------------
// SpectralProxLayer for MI455X (gfx1250, wave32, WMMA bf16 16x16x32)
//
// Heavy path: 3 chained GEMMs C = A(8192x8192) @ B(8192x256).
//   - A converted to bf16 once (128 MB ~ fits 192 MB L2 -> GEMM 2/3 hit L2)
//   - B kept in bf16 K-tiled layout Bt[k/32][n][k%32]: LDS staging is a
//     contiguous copy done with global_load_async_to_lds_b128 (ASYNCcnt)
//   - double-buffered LDS, 8 waves/block, wave tile 32x64, 16 WMMA/step
// The per-(k,s) 256x256 solve is eliminated via Sherman-Morrison:
//   (I + c phi phi^T)^{-1} phi = phi / (1 + c ||phi||^2)
// ---------------------------------------------------------------------------

#define NN   8192
#define DD   256
#define SS   32

typedef __attribute__((ext_vector_type(16))) __bf16 v16bf;
typedef __attribute__((ext_vector_type(8)))  __bf16 v8bf;
typedef __attribute__((ext_vector_type(8)))  float  v8f;

union Frag { v16bf v; v8bf h[2]; };

__device__ __forceinline__ unsigned int f2bf(float x) {
  unsigned int u = __float_as_uint(x);
  return (u + 0x7fffu + ((u >> 16) & 1u)) >> 16;   // round-to-nearest-even
}
__device__ __forceinline__ unsigned int pack2(float a, float b) {
  return f2bf(a) | (f2bf(b) << 16);
}

// gfx1250 async global->LDS copy (VGLOBAL GLOBAL_LOAD_ASYNC_TO_LDS_B128).
__device__ __forceinline__ void async_copy_b128(unsigned ldsAddr,
                                                const void* gaddr) {
  asm volatile("global_load_async_to_lds_b128 %0, %1, off"
               :: "v"(ldsAddr), "v"((unsigned long long)(uintptr_t)gaddr)
               : "memory");
}
__device__ __forceinline__ void wait_async0() {
  asm volatile("s_wait_asynccnt 0" ::: "memory");
}

// --------------------------- f32 -> bf16 convert (A) -----------------------
__global__ void cvt_bf16_kernel(const float* __restrict__ src,
                                unsigned short* __restrict__ dst, size_t n8) {
  size_t i = (size_t)blockIdx.x * blockDim.x + threadIdx.x;
  size_t stride = (size_t)gridDim.x * blockDim.x;
  const float4* s = (const float4*)src;
  uint4* d = (uint4*)dst;
  for (; i < n8; i += stride) {
    float4 f0 = s[i * 2 + 0];
    float4 f1 = s[i * 2 + 1];
    uint4 o;
    o.x = pack2(f0.x, f0.y);
    o.y = pack2(f0.z, f0.w);
    o.z = pack2(f1.x, f1.y);
    o.w = pack2(f1.z, f1.w);
    d[i] = o;
  }
}

// ---------------- f32 (8192x256) -> bf16 K-tiled Bt[k/32][n][k%32] ---------
__global__ __launch_bounds__(256)
void transpose_cvt_kernel(const float* __restrict__ src,
                          unsigned short* __restrict__ dst) {
  __shared__ float ld[32 * 257];
  const int t  = threadIdx.x;
  const int k0 = blockIdx.x * 32;
#pragma unroll
  for (int c = 0; c < 8; ++c) {
    int q   = c * 256 + t;              // 2048 float4 = 32 rows x 256 f32
    int row = q >> 6;
    int col = (q & 63) * 4;
    float4 f = *(const float4*)(src + (size_t)(k0 + row) * DD + col);
    ld[row * 257 + col + 0] = f.x;
    ld[row * 257 + col + 1] = f.y;
    ld[row * 257 + col + 2] = f.z;
    ld[row * 257 + col + 3] = f.w;
  }
  __syncthreads();
  uint32_t* d = (uint32_t*)dst + (size_t)blockIdx.x * 4096;
#pragma unroll
  for (int c = 0; c < 16; ++c) {
    int q  = c * 256 + t;               // 4096 dwords (8192 bf16)
    int n  = q >> 4;
    int kp = (q & 15) * 2;
    d[n * 16 + (q & 15)] = pack2(ld[kp * 257 + n], ld[(kp + 1) * 257 + n]);
  }
}

// --------------------------- bf16 WMMA GEMM --------------------------------
// C(8192x256) = A(8192x8192,bf16) @ B(8192x256,bf16 K-tiled), f32 accumulate.
// 256 thr = 8 waves; block tile 64(M)x256(N); wave tile 32x64.
// Epilogue also emits bf16 K-tiled C for the next chained GEMM.
#define LDST 40   // LDS row stride in bf16 units: 32 K + 8 pad (20-bank skew)

__global__ __launch_bounds__(256)
void gemm_bf16_wmma_kernel(const unsigned short* __restrict__ Ah,
                           const unsigned short* __restrict__ Bt,
                           float* __restrict__ C,
                           unsigned short* __restrict__ BtOut) {
  __shared__ alignas(16) unsigned short ldsA[2][64 * LDST];
  __shared__ alignas(16) unsigned short ldsB[2][256 * LDST];

  const int t     = threadIdx.x;
  const int lane  = t & 31;
  const int wave  = t >> 5;
  const int waveM = wave >> 2;           // 0..1
  const int nBase = (wave & 3) * 64;     // 0..192
  const int row16 = lane & 15;
  const int hihf  = lane >> 4;           // K-group select within fragment
  const int mBase = blockIdx.x * 64;

  v8f acc[2][4];
#pragma unroll
  for (int i = 0; i < 2; ++i)
#pragma unroll
    for (int j = 0; j < 4; ++j)
#pragma unroll
      for (int r = 0; r < 8; ++r) acc[i][j][r] = 0.0f;

  // async-stage one 64x32 A tile + 32x256 B tile into LDS buffer `buf`
  auto stage = [&](int buf, int k0) {
    unsigned aB = (unsigned)(uintptr_t)&ldsA[buf][0];
    unsigned bB = (unsigned)(uintptr_t)&ldsB[buf][0];
    {   // A: 64 rows x 64B -> 256 b128 chunks, 1 per thread
      int m = t >> 2, kc = t & 3;
      async_copy_b128(aB + m * (LDST * 2) + kc * 16,
                      Ah + (size_t)(mBase + m) * NN + k0 + kc * 8);
    }
#pragma unroll
    for (int c = 0; c < 4; ++c) {        // B: 256 rows x 64B -> 4 per thread
      int q = c * 256 + t;
      int n = q >> 2, kc = q & 3;
      async_copy_b128(bB + n * (LDST * 2) + kc * 16,
                      Bt + (size_t)k0 * DD + n * 32 + kc * 8);
    }
  };

  auto compute = [&](int buf) {
    Frag a[2];
#pragma unroll
    for (int i = 0; i < 2; ++i) {
      const unsigned short* p =
          &ldsA[buf][(waveM * 32 + i * 16 + row16) * LDST + hihf * 8];
      a[i].h[0] = *(const v8bf*)(p);
      a[i].h[1] = *(const v8bf*)(p + 16);
    }
#pragma unroll
    for (int j = 0; j < 4; ++j) {
      const unsigned short* p =
          &ldsB[buf][(nBase + j * 16 + row16) * LDST + hihf * 8];
      Frag b;
      b.h[0] = *(const v8bf*)(p);
      b.h[1] = *(const v8bf*)(p + 16);
#pragma unroll
      for (int i = 0; i < 2; ++i)
        acc[i][j] = __builtin_amdgcn_wmma_f32_16x16x32_bf16(
            false, a[i].v, false, b.v, (short)0, acc[i][j], false, false);
    }
  };

  stage(0, 0);
  wait_async0();
  __syncthreads();

  for (int k0 = 0; k0 < NN; k0 += 64) {  // 2 tiles per trip: buffers 0 and 1
    if (k0 + 64 < NN)
      __builtin_prefetch(Ah + (size_t)(mBase + (t & 63)) * NN + k0 + 64, 0, 0);
    if (k0 + 32 < NN) stage(1, k0 + 32);
    compute(0);
    if (k0 + 32 < NN) wait_async0();
    __syncthreads();
    if (k0 + 32 < NN) {
      if (k0 + 64 < NN) stage(0, k0 + 64);
      compute(1);
      if (k0 + 64 < NN) wait_async0();
      __syncthreads();
    }
  }

  // Epilogue: f32 C for consumers + bf16 K-tiled C as next GEMM's B
#pragma unroll
  for (int i = 0; i < 2; ++i)
#pragma unroll
    for (int j = 0; j < 4; ++j)
#pragma unroll
      for (int r = 0; r < 8; ++r) {
        int row = mBase + waveM * 32 + i * 16 + hihf * 8 + r;
        int col = nBase + j * 16 + row16;
        float v = acc[i][j][r];
        C[(size_t)row * DD + col] = v;
        BtOut[(size_t)(row >> 5) * (DD * 32) + col * 32 + (row & 31)] =
            (unsigned short)f2bf(v);
      }
}

// --------------------- small-weight kernel (softmax etc.) ------------------
// ws_w layout: [0..3]=hop_w  [4..35]=spec_w  [36..67]=lap_lams
__global__ void weights_kernel(const float* __restrict__ hop_weights,
                               const float* __restrict__ spec_logits,
                               const float* __restrict__ lap_logits,
                               float* __restrict__ ws_w,
                               float* __restrict__ out_tail) {
  int lane = threadIdx.x;  // 32 threads
  float x = spec_logits[lane];
  float m = x;
#pragma unroll
  for (int k = 16; k; k >>= 1) m = fmaxf(m, __shfl_xor(m, k, 32));
  float e = __expf(x - m);
  float s = e;
#pragma unroll
  for (int k = 16; k; k >>= 1) s += __shfl_xor(s, k, 32);
  float sw = e / s;
  ws_w[4 + lane]     = sw;
  out_tail[4 + lane] = sw;                 // spec_w output (after hop_w)

  float l  = lap_logits[lane];
  float sp = (l > 20.0f) ? l : log1pf(__expf(l));
  ws_w[36 + lane] = sp;

  if (lane == 0) {
    float h0 = hop_weights[0], h1 = hop_weights[1];
    float h2 = hop_weights[2], h3 = hop_weights[3];
    float mm = fmaxf(fmaxf(h0, h1), fmaxf(h2, h3));
    float e0 = __expf(h0 - mm), e1 = __expf(h1 - mm);
    float e2 = __expf(h2 - mm), e3 = __expf(h3 - mm);
    float ss = e0 + e1 + e2 + e3;
    ws_w[0] = e0 / ss; ws_w[1] = e1 / ss; ws_w[2] = e2 / ss; ws_w[3] = e3 / ss;
    out_tail[0] = e0 / ss; out_tail[1] = e1 / ss;
    out_tail[2] = e2 / ss; out_tail[3] = e3 / ss;
  }
}

// ----------------------------- Phi = V^T @ hop -----------------------------
__global__ __launch_bounds__(256)
void phi_kernel(const float* __restrict__ V, const float* __restrict__ h0,
                const float* __restrict__ h1, const float* __restrict__ h2,
                const float* __restrict__ h3, float* __restrict__ Phi) {
  const int k = blockIdx.x;
  const float* Hk = (k == 0) ? h0 : (k == 1) ? h1 : (k == 2) ? h2 : h3;
  const int dBase = blockIdx.y * 64;
  const int t  = threadIdx.x;
  const int dI = t & 63;
  const int sg = t >> 6;  // 4 groups of 8 spectral components
  __shared__ float Vl[64 * SS];

  float acc[8];
#pragma unroll
  for (int j = 0; j < 8; ++j) acc[j] = 0.0f;

  for (int nc = 0; nc < NN; nc += 64) {
#pragma unroll
    for (int r = 0; r < 8; ++r) {
      int idx = t + 256 * r;             // 2048 floats
      Vl[idx] = V[(size_t)nc * SS + idx];
    }
    __syncthreads();
    for (int nn = 0; nn < 64; ++nn) {
      float hv = Hk[(size_t)(nc + nn) * DD + dBase + dI];
#pragma unroll
      for (int j = 0; j < 8; ++j) acc[j] += Vl[nn * SS + sg * 8 + j] * hv;
    }
    __syncthreads();
  }
#pragma unroll
  for (int j = 0; j < 8; ++j) {
    int s = sg * 8 + j;
    Phi[((size_t)k * SS + s) * DD + dBase + dI] = acc[j];
  }
}

// --------------- T = ETA*(T_R - T_L) via Sherman-Morrison ------------------
__global__ __launch_bounds__(256)
void t_kernel(const float* __restrict__ Phi, const float* __restrict__ ws_w,
              const float* __restrict__ eigvals, float* __restrict__ T) {
  const int s = blockIdx.x;
  const int d = threadIdx.x;
  __shared__ float red[256];

  float phiv[4], nrm[4];
#pragma unroll
  for (int k = 0; k < 4; ++k)
    phiv[k] = Phi[((size_t)k * SS + s) * DD + d];
#pragma unroll
  for (int k = 0; k < 4; ++k) {
    red[d] = phiv[k] * phiv[k];
    __syncthreads();
    for (int off = 128; off; off >>= 1) {
      if (d < off) red[d] += red[d + off];
      __syncthreads();
    }
    nrm[k] = red[0];
    __syncthreads();
  }
  const float coeff = 256.0f / (8192.0f * 0.25f);  // d/(n*EPS^2) = 0.125
  float sw  = ws_w[4 + s];
  float lam = ws_w[36 + s] * eigvals[s];
  float acc = 0.0f;
#pragma unroll
  for (int k = 0; k < 4; ++k)
    acc += ws_w[k] * phiv[k] * (sw * coeff / (1.0f + coeff * nrm[k]) - lam);
  T[(size_t)s * DD + d] = 0.5f * acc;  // ETA
}

// --------- fused: H + V@T -> soft-threshold -> LayerNorm (wave32) ----------
__global__ __launch_bounds__(256)
void final_kernel(const float* __restrict__ H, const float* __restrict__ V,
                  const float* __restrict__ T, const float* __restrict__ thr,
                  const float* __restrict__ gamma,
                  const float* __restrict__ beta, float* __restrict__ out) {
  __shared__ float Tl[SS * DD];
  const int t = threadIdx.x;
#pragma unroll
  for (int c = 0; c < 32; ++c) Tl[c * DD + t] = T[(size_t)c * DD + t];
  __syncthreads();

  const int lane = t & 31;
  const int wave = t >> 5;
  const int base = blockIdx.x * 128 + wave * 16;

  for (int i = 0; i < 16; ++i) {
    int n = base + i;
    float vv = V[(size_t)n * SS + lane];
    float g[8];
#pragma unroll
    for (int j = 0; j < 8; ++j) g[j] = 0.0f;
    for (int s = 0; s < SS; ++s) {
      float vs = __shfl(vv, s, 32);
#pragma unroll
      for (int j = 0; j < 8; ++j) g[j] += vs * Tl[s * DD + lane + 32 * j];
    }
    float st[8], sum = 0.0f, sq = 0.0f;
#pragma unroll
    for (int j = 0; j < 8; ++j) {
      int d = lane + 32 * j;
      float h = H[(size_t)n * DD + d] + g[j];
      float a = fabsf(h) - thr[d];
      a = a > 0.0f ? a : 0.0f;
      float v2 = copysignf(a, h);
      st[j] = v2;
      sum += v2;
      sq  += v2 * v2;
    }
#pragma unroll
    for (int k = 16; k; k >>= 1) {
      sum += __shfl_xor(sum, k, 32);
      sq  += __shfl_xor(sq, k, 32);
    }
    float mu  = sum * (1.0f / 256.0f);
    float var = sq * (1.0f / 256.0f) - mu * mu;
    float rs  = rsqrtf(var + 1e-5f);
#pragma unroll
    for (int j = 0; j < 8; ++j) {
      int d = lane + 32 * j;
      out[(size_t)n * DD + d] = (st[j] - mu) * rs * gamma[d] + beta[d];
    }
  }
}

// ---------------------------------------------------------------------------
extern "C" void kernel_launch(void* const* d_in, const int* in_sizes, int n_in,
                              void* d_out, int out_size, void* d_ws,
                              size_t ws_size, hipStream_t stream) {
  const float* H     = (const float*)d_in[0];
  const float* A     = (const float*)d_in[1];
  const float* V     = (const float*)d_in[2];
  const float* eig   = (const float*)d_in[3];
  const float* specl = (const float*)d_in[4];
  const float* lapl  = (const float*)d_in[5];
  const float* hopw  = (const float*)d_in[6];
  const float* thr   = (const float*)d_in[7];
  const float* gam   = (const float*)d_in[8];
  const float* bet   = (const float*)d_in[9];
  float* out = (float*)d_out;

  const size_t HOP = (size_t)NN * DD;           // 2,097,152 elements
  char* ws = (char*)d_ws;
  // workspace layout (~168 MB total)
  unsigned short* Abf = (unsigned short*)ws;                          // 128 MB
  unsigned short* Bt0 = (unsigned short*)(ws + (size_t)NN * NN * 2);  // 4 MB
  unsigned short* Bt1 = Bt0 + HOP;                                    // 4 MB
  float* hop1 = (float*)(Bt1 + HOP);
  float* hop2 = hop1 + HOP;
  float* hop3 = hop2 + HOP;
  float* PhiP = hop3 + HOP;               // 4*32*256
  float* Tp   = PhiP + 4 * SS * DD;       // 32*256
  float* wsw  = Tp + SS * DD;             // 68 floats

  // small weights (also writes hop_w/spec_w tail of d_out)
  weights_kernel<<<1, 32, 0, stream>>>(hopw, specl, lapl, wsw, out + HOP);

  // A -> bf16 once (128 MB, ~L2-resident for GEMMs 2 and 3)
  cvt_bf16_kernel<<<8192, 256, 0, stream>>>(A, Abf, (size_t)NN * NN / 8);

  // H -> bf16 K-tiled (first B); chained GEMMs regenerate tiled B themselves
  transpose_cvt_kernel<<<NN / 32, 256, 0, stream>>>(H, Bt0);
  gemm_bf16_wmma_kernel<<<NN / 64, 256, 0, stream>>>(Abf, Bt0, hop1, Bt1);
  gemm_bf16_wmma_kernel<<<NN / 64, 256, 0, stream>>>(Abf, Bt1, hop2, Bt0);
  gemm_bf16_wmma_kernel<<<NN / 64, 256, 0, stream>>>(Abf, Bt0, hop3, Bt1);

  // Phi[k,s,d] = sum_n V[n,s] * hop_k[n,d]
  phi_kernel<<<dim3(4, 4), 256, 0, stream>>>(V, H, hop1, hop2, hop3, PhiP);

  // T = ETA * (spec_w * hop_w-weighted q  -  lam*eig * hop_w-weighted Phi)
  t_kernel<<<SS, 256, 0, stream>>>(PhiP, wsw, eig, Tp);

  // H_out = LN(softthresh(H + V @ T))
  final_kernel<<<NN / 128, 256, 0, stream>>>(H, V, Tp, thr, gam, bet, out);
}
